// VectorQuantizer_24369644438016
// MI455X (gfx1250) — matrix operand order repaired and checked by reading, not compiled, not run on hardware
//
#include <hip/hip_runtime.h>
#include <hip/hip_bf16.h>

typedef __bf16 v16bf __attribute__((ext_vector_type(16)));
typedef float  v8f   __attribute__((ext_vector_type(8)));
typedef unsigned int u32x4 __attribute__((ext_vector_type(4)));
typedef int          i32x4 __attribute__((ext_vector_type(4)));
typedef int          i32x8 __attribute__((ext_vector_type(8)));

#define N_CODES 16384
#define E_DIM   256
#define NBATCH  8
#define HW      1024
#define M_TOT   8192            // NBATCH * HW
#define ZQ_N    2097152         // 8*256*32*32
#define NSPLIT  8
#define BLOCK_M 128
#define NCHUNK  (N_CODES / NSPLIT)
#define LDSTRIDE 528            // 256 bf16 = 512B, +16B pad vs LDS bank conflicts

// ---- workspace layout (byte offsets; all 16B aligned) ----
#define WS_ZF_HI   0
#define WS_ZF_LO   (WS_ZF_HI  + M_TOT*E_DIM*2)
#define WS_EMB_HI  (WS_ZF_LO  + M_TOT*E_DIM*2)
#define WS_EMB_LO  (WS_EMB_HI + N_CODES*E_DIM*2)
#define WS_CNORM   (WS_EMB_LO + N_CODES*E_DIM*2)
#define WS_PVAL    (WS_CNORM  + N_CODES*4)
#define WS_PIDX    (WS_PVAL   + M_TOT*NSPLIT*4)
#define WS_IDXF    (WS_PIDX   + M_TOT*NSPLIT*4)
#define WS_LOSSP   (WS_IDXF   + M_TOT*4)      // 2048 floats

union UFrag { uint4 q[2]; v16bf v; };

// ---------------- TDM: 2-D global->LDS tile copy with 16B/row LDS padding ----
// D# per CDNA5 ISA ch.8: group0 = {count/lds_addr/global_addr/type},
// group1 = {flags, dims, tile dims, strides}; groups 2/3 unused (2-D tensor).
// Copies BLOCK_M rows of 512B, inserting 16B LDS padding after every 512B
// (pad_interval=6 -> 64 x 8B = 512B, pad_amount=3 -> 4 DWORDs = 16B).
__device__ static inline void tdm_load_rows_padded(const void* gsrc, unsigned ldsOff) {
#if __has_builtin(__builtin_amdgcn_tensor_load_to_lds)
    unsigned long long ga = (unsigned long long)gsrc;
    u32x4 g0;
    g0[0] = 0x1u;                                   // count=1 (valid user D#)
    g0[1] = ldsOff;                                 // lds_addr (bytes)
    g0[2] = (unsigned)(ga & 0xffffffffu);           // global_addr[31:0]
    g0[3] = (unsigned)((ga >> 32) & 0x01ffffffu)    // global_addr[56:32]
          | 0x80000000u;                            // type=2 ("image")
    i32x8 g1;
    g1[0] = (int)((3u << 16)      // data_size = 8B units
                | (1u << 20)      // pad_enable
                | (6u << 22)      // pad_interval: 64 x 8B = 512B
                | (3u << 25));    // pad_amount:   4 DWORDs = 16B
    g1[1] = (int)((64u & 0xffffu) << 16);                       // tensor_dim0 = 64 lo
    g1[2] = (int)(((unsigned)M_TOT & 0xffffu) << 16);           // dim0 hi=0 | tensor_dim1 lo
    g1[3] = (int)((((unsigned)M_TOT >> 16) & 0xffffu) | (64u << 16)); // dim1 hi | tile_dim0=64
    g1[4] = (int)((unsigned)BLOCK_M & 0xffffu);                 // tile_dim1=128, tile_dim2=0
    g1[5] = 64;                                                 // tensor_dim0_stride = 64 (8B units)
    g1[6] = 0;
    g1[7] = 0;
    i32x4 z4 = {0, 0, 0, 0};
#if __has_include(<hip/amd_detail/amd_gfx1250_TDM.h>)
    i32x8 z8 = {0, 0, 0, 0, 0, 0, 0, 0};
    __builtin_amdgcn_tensor_load_to_lds(g0, g1, z4, z4, z8, 0);   // clang-23 arity
#else
    __builtin_amdgcn_tensor_load_to_lds(g0, g1, z4, z4, 0);       // ROCm 7.2 arity
#endif
#else
    (void)gsrc; (void)ldsOff;
#endif
}

// ---------------- kernel 1: codebook -> bf16 hi/lo + ||e||^2 ----------------
__global__ void k_prep_codebook(const float* __restrict__ emb,
                                __bf16* __restrict__ ehi, __bf16* __restrict__ elo,
                                float* __restrict__ cnorm) {
    int wave = threadIdx.x >> 5;
    int lane = threadIdx.x & 31;
    int row  = blockIdx.x * 8 + wave;

    const float4* src = (const float4*)(emb + row * E_DIM + lane * 8);
    float4 f0 = src[0], f1 = src[1];
    float v[8] = {f0.x, f0.y, f0.z, f0.w, f1.x, f1.y, f1.z, f1.w};

    union { __bf16 b[8]; uint4 q; } uh, ul;
    float s = 0.f;
#pragma unroll
    for (int i = 0; i < 8; i++) {
        float x = v[i];
        s += x * x;
        __bf16 hh = (__bf16)x;
        uh.b[i] = hh;
        ul.b[i] = (__bf16)(x - (float)hh);
    }
    *(uint4*)(ehi + row * E_DIM + lane * 8) = uh.q;
    *(uint4*)(elo + row * E_DIM + lane * 8) = ul.q;

#pragma unroll
    for (int d = 16; d > 0; d >>= 1) s += __shfl_xor(s, d, 32);
    if (lane == 0) cnorm[row] = s;
}

// ---------------- kernel 2: NCHW z -> zf[M][K] bf16 hi/lo (LDS transpose) ----
__global__ void k_prep_z(const float* __restrict__ z,
                         __bf16* __restrict__ zhi, __bf16* __restrict__ zlo) {
    __shared__ float tile[32][33];
    int b   = blockIdx.z;
    int c0  = blockIdx.y * 32;
    int hw0 = blockIdx.x * 32;
    int tx  = threadIdx.x & 31, ty = threadIdx.x >> 5;

#pragma unroll
    for (int r = 0; r < 4; r++) {
        int cl = ty + 8 * r;
        tile[cl][tx] = z[((b * E_DIM + c0 + cl) * HW) + hw0 + tx];
    }
    __syncthreads();
#pragma unroll
    for (int r = 0; r < 4; r++) {
        int hwl = ty + 8 * r;
        int m   = b * HW + hw0 + hwl;
        float x = tile[tx][hwl];
        __bf16 hh = (__bf16)x;
        zhi[m * E_DIM + c0 + tx] = hh;
        zlo[m * E_DIM + c0 + tx] = (__bf16)(x - (float)hh);
    }
}

// ---------------- kernel 3: fused GEMM + argmin (WMMA bf16 3-term split) ----
__launch_bounds__(256)
__global__ void k_vq_argmin(const __bf16* __restrict__ zhi, const __bf16* __restrict__ zlo,
                            const __bf16* __restrict__ ehi, const __bf16* __restrict__ elo,
                            const float* __restrict__ cnorm,
                            float* __restrict__ pval, int* __restrict__ pidx) {
    extern __shared__ char smem[];
    char* sAhi = smem;
    char* sAlo = smem + BLOCK_M * LDSTRIDE;

    const int tid = threadIdx.x;
    const int m0  = blockIdx.x * BLOCK_M;
    const int n0  = blockIdx.y * NCHUNK;

    // stage 128 z-rows (hi+lo) into LDS once, with 16B per-row padding
#if __has_builtin(__builtin_amdgcn_tensor_load_to_lds)
    if (tid < 32) {   // one wave issues the TDM descriptors (TDM ignores EXEC)
        tdm_load_rows_padded(zhi + (size_t)m0 * E_DIM, (unsigned)(size_t)sAhi);
        tdm_load_rows_padded(zlo + (size_t)m0 * E_DIM, (unsigned)(size_t)sAlo);
        __builtin_amdgcn_s_wait_tensorcnt(0);
    }
#else
    for (int i = tid; i < BLOCK_M * 32; i += 256) {
        int row = i >> 5, c16 = i & 31;
        *(uint4*)(sAhi + row * LDSTRIDE + c16 * 16) =
            *(const uint4*)(zhi + (m0 + row) * E_DIM + c16 * 8);
        *(uint4*)(sAlo + row * LDSTRIDE + c16 * 16) =
            *(const uint4*)(zlo + (m0 + row) * E_DIM + c16 * 8);
    }
#endif
    __syncthreads();

    const int wave = tid >> 5, lane = tid & 31;
    const int ph   = lane >> 4;       // which 16-lane half
    const int col  = lane & 15;

    // A fragments: wave owns rows [16*wave, 16*wave+16), full K=256 in registers
    v16bf ahi[8], alo[8];
    {
        const char* rh = sAhi + (wave * 16 + col) * LDSTRIDE;
        const char* rl = sAlo + (wave * 16 + col) * LDSTRIDE;
#pragma unroll
        for (int s = 0; s < 8; s++) {
            int off1 = 2 * (32 * s + 8 * ph);   // K = 32s + 8*ph .. +8
            UFrag u;
            u.q[0] = *(const uint4*)(rh + off1);
            u.q[1] = *(const uint4*)(rh + off1 + 32);  // K = 32s + 16 + 8*ph
            ahi[s] = u.v;
            u.q[0] = *(const uint4*)(rl + off1);
            u.q[1] = *(const uint4*)(rl + off1 + 32);
            alo[s] = u.v;
        }
    }

    float best[8];
    int   bidx[8];
#pragma unroll
    for (int r = 0; r < 8; r++) { best[r] = 3.4e38f; bidx[r] = 0x7fffffff; }

    for (int nt = n0; nt < n0 + NCHUNK; nt += 16) {
        const int n = nt + col;                 // this lane's codebook row
        const char* bh = (const char*)(ehi + n * E_DIM + 8 * ph);
        const char* bl = (const char*)(elo + n * E_DIM + 8 * ph);
        __builtin_prefetch((const void*)(bh + 16 * E_DIM * 2), 0, 3);
        __builtin_prefetch((const void*)(bl + 16 * E_DIM * 2), 0, 3);

        v8f acc0 = {}, acc1 = {};               // two chains: halve D->C dependency depth
#pragma unroll
        for (int s = 0; s < 8; s++) {
            UFrag ub, vb;
            ub.q[0] = *(const uint4*)(bh + s * 64);
            ub.q[1] = *(const uint4*)(bh + s * 64 + 32);
            vb.q[0] = *(const uint4*)(bl + s * 64);
            vb.q[1] = *(const uint4*)(bl + s * 64 + 32);
            acc0 = __builtin_amdgcn_wmma_f32_16x16x32_bf16(false, ahi[s], false, ub.v,
                                                           (short)0, acc0, false, false);
            acc1 = __builtin_amdgcn_wmma_f32_16x16x32_bf16(false, ahi[s], false, vb.v,
                                                           (short)0, acc1, false, false);
            acc0 = __builtin_amdgcn_wmma_f32_16x16x32_bf16(false, alo[s], false, ub.v,
                                                           (short)0, acc0, false, false);
        }

        const float cn = cnorm[n];
#pragma unroll
        for (int r = 0; r < 8; r++) {
            float v = cn - 2.0f * (acc0[r] + acc1[r]);  // dist - ||z||^2 (row-const dropped)
            if (v < best[r]) { best[r] = v; bidx[r] = n; }
        }
    }

    // reduce (val,idx) across the 16 lanes of each half; tie -> lower index
#pragma unroll
    for (int r = 0; r < 8; r++) {
        float v = best[r];
        int   i = bidx[r];
#pragma unroll
        for (int d = 1; d < 16; d <<= 1) {
            float ov = __shfl_xor(v, d, 32);
            int   oi = __shfl_xor(i, d, 32);
            if (ov < v || (ov == v && oi < i)) { v = ov; i = oi; }
        }
        if (col == 0) {
            int mrow = m0 + wave * 16 + ph * 8 + r;   // C layout: VGPR r -> M = r + 8*ph
            pval[mrow * NSPLIT + blockIdx.y] = v;
            pidx[mrow * NSPLIT + blockIdx.y] = i;
        }
    }
}

// ---------------- kernel 4: merge N-splits, emit indices ----------------
__global__ void k_merge(const float* __restrict__ pval, const int* __restrict__ pidx,
                        int* __restrict__ idxf, float* __restrict__ out) {
    int m = blockIdx.x * 256 + threadIdx.x;
    float bv = 3.4e38f;
    int   bi = 0x7fffffff;
#pragma unroll
    for (int j = 0; j < NSPLIT; j++) {
        float v = pval[m * NSPLIT + j];
        int   i = pidx[m * NSPLIT + j];
        if (v < bv || (v == bv && i < bi)) { bv = v; bi = i; }
    }
    idxf[m] = bi;
    out[ZQ_N + 1 + m] = (float)bi;
}

// ---------------- kernel 5: gather + NCHW write + loss partials ----------------
__global__ void k_gather(const float* __restrict__ emb, const float* __restrict__ z,
                         const int* __restrict__ idxf, float* __restrict__ out,
                         float* __restrict__ lossp) {
    __shared__ float tile[32][33];
    __shared__ float wsum[8];
    int b   = blockIdx.z;
    int c0  = blockIdx.y * 32;
    int hw0 = blockIdx.x * 32;
    int tx  = threadIdx.x & 31, ty = threadIdx.x >> 5;

#pragma unroll
    for (int r = 0; r < 4; r++) {
        int hwl = ty + 8 * r;
        int m   = b * HW + hw0 + hwl;
        tile[hwl][tx] = emb[idxf[m] * E_DIM + c0 + tx];   // coalesced row gather
    }
    __syncthreads();

    float ls = 0.f;
#pragma unroll
    for (int r = 0; r < 4; r++) {
        int cl   = ty + 8 * r;
        int addr = (b * E_DIM + c0 + cl) * HW + hw0 + tx; // coalesced NCHW
        float q  = tile[tx][cl];
        out[addr] = q;                                    // straight-through z_q
        float d = q - z[addr];
        ls += d * d;
    }
#pragma unroll
    for (int d = 16; d > 0; d >>= 1) ls += __shfl_xor(ls, d, 32);
    if (tx == 0) wsum[ty] = ls;
    __syncthreads();
    if (threadIdx.x == 0) {
        float t = 0.f;
#pragma unroll
        for (int i = 0; i < 8; i++) t += wsum[i];
        lossp[(blockIdx.z * 8 + blockIdx.y) * 32 + blockIdx.x] = t;
    }
}

// ---------------- kernel 6: deterministic loss reduction ----------------
__global__ void k_loss(const float* __restrict__ lossp, float* __restrict__ out) {
    __shared__ float sb[256];
    float s = 0.f;
#pragma unroll
    for (int j = 0; j < 8; j++) s += lossp[threadIdx.x + 256 * j];
    sb[threadIdx.x] = s;
    __syncthreads();
    for (int st = 128; st > 0; st >>= 1) {
        if (threadIdx.x < st) sb[threadIdx.x] += sb[threadIdx.x + st];
        __syncthreads();
    }
    // loss = beta*mean + mean = 2*mean((z_q - z)^2)
    if (threadIdx.x == 0) out[ZQ_N] = 2.0f * sb[0] / (float)ZQ_N;
}

extern "C" void kernel_launch(void* const* d_in, const int* in_sizes, int n_in,
                              void* d_out, int out_size, void* d_ws, size_t ws_size,
                              hipStream_t stream) {
    const float* z   = (const float*)d_in[0];
    const float* emb = (const float*)d_in[1];
    float* out = (float*)d_out;
    char*  ws  = (char*)d_ws;

    __bf16* zhi  = (__bf16*)(ws + WS_ZF_HI);
    __bf16* zlo  = (__bf16*)(ws + WS_ZF_LO);
    __bf16* ehi  = (__bf16*)(ws + WS_EMB_HI);
    __bf16* elo  = (__bf16*)(ws + WS_EMB_LO);
    float*  cnrm = (float*)(ws + WS_CNORM);
    float*  pval = (float*)(ws + WS_PVAL);
    int*    pidx = (int*)(ws + WS_PIDX);
    int*    idxf = (int*)(ws + WS_IDXF);
    float*  losp = (float*)(ws + WS_LOSSP);

    k_prep_codebook<<<N_CODES / 8, 256, 0, stream>>>(emb, ehi, elo, cnrm);
    k_prep_z<<<dim3(32, 8, 8), 256, 0, stream>>>(z, zhi, zlo);

    size_t smem = (size_t)2 * BLOCK_M * LDSTRIDE;   // 132 KB of the 320 KB WGP LDS
    k_vq_argmin<<<dim3(M_TOT / BLOCK_M, NSPLIT), 256, smem, stream>>>(
        zhi, zlo, ehi, elo, cnrm, pval, pidx);

    k_merge<<<M_TOT / 256, 256, 0, stream>>>(pval, pidx, idxf, out);
    k_gather<<<dim3(32, 8, 8), 256, 0, stream>>>(emb, z, idxf, out, losp);
    k_loss<<<1, 256, 0, stream>>>(losp, out);
}
